// dynedge_energy_14336600834595
// MI455X (gfx1250) — compile-verified, matrix operand order checked by reading, not run on hardware
//
#include <hip/hip_runtime.h>

typedef _Float16 h16;
typedef __attribute__((ext_vector_type(16))) _Float16 v16h;
typedef __attribute__((ext_vector_type(8)))  _Float16 v8h;
typedef __attribute__((ext_vector_type(8)))  float    v8f;

#define LEAKY(x) ((x) >= 0.f ? (x) : 0.01f * (x))

// ---------------------------------------------------------------------------
// WMMA 16-bit fragment load (ISA 7.12.2). p = row start (row=M for A, N for
// Bt) already offset by kb = (lane>=16 ? 8 : 0). VGPR0..3 hold K pairs
// kb+0..7, VGPR4..7 hold kb+16..23 -> two aligned 16B loads.
// ---------------------------------------------------------------------------
__device__ __forceinline__ v16h frag_load(const h16* p) {
  v16h f;
  *(v8h*)&f       = *(const v8h*)p;
  *((v8h*)&f + 1) = *(const v8h*)(p + 16);
  return f;
}
__device__ __forceinline__ v16h frag_load_nt(const h16* p) {
  v16h f;
  *(v8h*)&f       = __builtin_nontemporal_load((const v8h*)p);
  *((v8h*)&f + 1) = __builtin_nontemporal_load((const v8h*)(p + 16));
  return f;
}

// ---------------------------------------------------------------------------
// Weight prep: W (K x N, f32 row-major) -> Wt (Npad x Kpad, f16, zero pad),
// bias (N) -> bp (Npad, zero pad).
// ---------------------------------------------------------------------------
__global__ void prep_weight(const float* __restrict__ W, const float* __restrict__ b,
                            h16* __restrict__ Wt, float* __restrict__ bp,
                            int K, int N, int Kpad, int Npad) {
  size_t gid = (size_t)blockIdx.x * blockDim.x + threadIdx.x;
  size_t total = (size_t)Npad * Kpad;
  if (gid < total) {
    int n = (int)(gid / Kpad);
    int k = (int)(gid % Kpad);
    float v = (n < N && k < K) ? W[(size_t)k * N + n] : 0.f;
    Wt[gid] = (h16)v;
  }
  if (gid < (size_t)Npad) bp[gid] = (gid < (size_t)N) ? b[gid] : 0.f;
}

// ---------------------------------------------------------------------------
// KNN: one block = 256 points of one batch; whole batch's xyz cached in LDS.
// ---------------------------------------------------------------------------
__global__ void knn_kernel(const float* __restrict__ P, int* __restrict__ idx,
                           int nper, int F) {
  __shared__ float pos[2048 * 3];
  int b = blockIdx.y;
  for (int t = threadIdx.x; t < nper; t += blockDim.x) {
    const float* p = P + ((size_t)b * nper + t) * F;
    pos[t * 3 + 0] = p[0];
    pos[t * 3 + 1] = p[1];
    pos[t * 3 + 2] = p[2];
  }
  __syncthreads();
  int i = blockIdx.x * blockDim.x + threadIdx.x;
  if (i >= nper) return;
  float px = pos[i * 3], py = pos[i * 3 + 1], pz = pos[i * 3 + 2];
  float bd[8];
  int bi[8];
#pragma unroll
  for (int k = 0; k < 8; ++k) { bd[k] = 3.4e38f; bi[k] = 0; }
  for (int j = 0; j < nper; ++j) {
    if (j == i) continue;
    float dx = pos[j * 3] - px, dy = pos[j * 3 + 1] - py, dz = pos[j * 3 + 2] - pz;
    float d2 = dx * dx + dy * dy + dz * dz;
    if (d2 < bd[7]) {
      int p = 7;
      while (p > 0 && d2 < bd[p - 1]) { bd[p] = bd[p - 1]; bi[p] = bi[p - 1]; --p; }
      bd[p] = d2; bi[p] = j;
    }
  }
  int* o = idx + ((size_t)b * nper + i) * 8;
#pragma unroll
  for (int k = 0; k < 8; ++k) o[k] = bi[k];
}

// ---------------------------------------------------------------------------
// Build edge features: row e = (point p, neighbor k); cols [xi, xj-xi, pad0],
// f32 -> f16.
// ---------------------------------------------------------------------------
__global__ void build_edge(const float* __restrict__ P, const int* __restrict__ idx,
                           h16* __restrict__ E, int nper, int F, int ldE, long long NE) {
  size_t gid = (size_t)blockIdx.x * blockDim.x + threadIdx.x;
  size_t total = (size_t)NE * ldE;
  if (gid >= total) return;
  int c = (int)(gid % ldE);
  size_t e = gid / ldE;
  size_t p = e >> 3;  // global point index b*nper+i
  float v = 0.f;
  if (c < 2 * F) {
    size_t b = p / (size_t)nper;
    int j = idx[e];
    const float* xi = P + p * F;
    const float* xj = P + (b * nper + j) * F;
    v = (c < F) ? xi[c] : (xj[c - F] - xi[c - F]);
  }
  E[gid] = (h16)v;
}

// ---------------------------------------------------------------------------
// Build concat [x(8) | a(192) | b(192) | c(192) | d(192) | pad] -> f16, ld=800
// ---------------------------------------------------------------------------
__global__ void build_concat(const float* __restrict__ x, const float* __restrict__ a,
                             const float* __restrict__ bm, const float* __restrict__ cm,
                             const float* __restrict__ dm, h16* __restrict__ C,
                             int ldc, long long NP) {
  size_t gid = (size_t)blockIdx.x * blockDim.x + threadIdx.x;
  size_t total = (size_t)NP * ldc;
  if (gid >= total) return;
  int c = (int)(gid % ldc);
  size_t p = gid / ldc;
  float v = 0.f;
  if (c < 8)        v = x[p * 8 + c];
  else if (c < 200) v = a[p * 192 + (c - 8)];
  else if (c < 392) v = bm[p * 192 + (c - 200)];
  else if (c < 584) v = cm[p * 192 + (c - 392)];
  else if (c < 776) v = dm[p * 192 + (c - 584)];
  C[gid] = (h16)v;
}

// ---------------------------------------------------------------------------
// WMMA GEMM, f16 out: O = act(A @ Bt^T + bias). A: M x KPAD f16 rowmajor,
// Bt: Npad x KPAD f16 (N-major), O: M x Npad f16.
// One wave = MT*16 rows x NT*16 cols; MT*NT independent WMMAs per K-step.
// KPAD is compile-time so all row offsets fold into load immediates (2 base
// pointers total). Tile sizes chosen to stay spill-free (~130 VGPRs).
// ---------------------------------------------------------------------------
template <int MT, int NT, int KPAD>
__global__ void wmma_gemm_f16out(const h16* __restrict__ A, const h16* __restrict__ Bt,
                                 const float* __restrict__ bias, h16* __restrict__ O,
                                 int M, int Npad, int strips, int act) {
  int wave = blockIdx.x * (blockDim.x >> 5) + (threadIdx.x >> 5);
  int mgroups = M / (16 * MT);
  if (wave >= mgroups * strips) return;
  int mg = wave / strips, ct = wave % strips;
  int lane = threadIdx.x & 31;
  int nloc = lane & 15;
  int hi = lane >> 4;
  int kb = hi ? 8 : 0;

  const h16* a0 = A + ((size_t)(mg * MT * 16 + nloc)) * KPAD + kb;
  const h16* b0 = Bt + ((size_t)(ct * (NT * 16) + nloc)) * KPAD + kb;
  v8f acc[MT][NT];
#pragma unroll
  for (int m = 0; m < MT; ++m)
#pragma unroll
    for (int t = 0; t < NT; ++t) acc[m][t] = (v8f){};

  for (int k0 = 0; k0 < KPAD; k0 += 32) {
    v16h af[MT], bf[NT];
#pragma unroll
    for (int m = 0; m < MT; ++m) af[m] = frag_load_nt(a0 + m * 16 * KPAD + k0);
#pragma unroll
    for (int t = 0; t < NT; ++t) bf[t] = frag_load(b0 + t * 16 * KPAD + k0);
#pragma unroll
    for (int m = 0; m < MT; ++m)
#pragma unroll
      for (int t = 0; t < NT; ++t)
        acc[m][t] = __builtin_amdgcn_wmma_f32_16x16x32_f16(false, af[m], false, bf[t],
                                                           (short)0, acc[m][t], false, false);
  }
#pragma unroll
  for (int m = 0; m < MT; ++m) {
#pragma unroll
    for (int t = 0; t < NT; ++t) {
      int col = ct * (NT * 16) + t * 16 + nloc;
      float bv = bias[col];
      h16* op = O + ((size_t)((mg * MT + m) * 16 + hi * 8)) * Npad + col;
#pragma unroll
      for (int r = 0; r < 8; ++r) {
        float x = acc[m][t][r] + bv;
        if (act) x = LEAKY(x);
        op[(size_t)r * Npad] = (h16)x;
      }
    }
  }
}

// ---------------------------------------------------------------------------
// WMMA GEMM, f32 out: O = act(A @ Bt^T + bias), O: M x Npad f32.
// ---------------------------------------------------------------------------
template <int MT, int NT, int KPAD>
__global__ void wmma_gemm_f32out(const h16* __restrict__ A, const h16* __restrict__ Bt,
                                 const float* __restrict__ bias, float* __restrict__ O,
                                 int M, int Npad, int strips, int act) {
  int wave = blockIdx.x * (blockDim.x >> 5) + (threadIdx.x >> 5);
  int mgroups = M / (16 * MT);
  if (wave >= mgroups * strips) return;
  int mg = wave / strips, ct = wave % strips;
  int lane = threadIdx.x & 31;
  int nloc = lane & 15;
  int hi = lane >> 4;
  int kb = hi ? 8 : 0;

  const h16* a0 = A + ((size_t)(mg * MT * 16 + nloc)) * KPAD + kb;
  const h16* b0 = Bt + ((size_t)(ct * (NT * 16) + nloc)) * KPAD + kb;
  v8f acc[MT][NT];
#pragma unroll
  for (int m = 0; m < MT; ++m)
#pragma unroll
    for (int t = 0; t < NT; ++t) acc[m][t] = (v8f){};

  for (int k0 = 0; k0 < KPAD; k0 += 32) {
    v16h af[MT], bf[NT];
#pragma unroll
    for (int m = 0; m < MT; ++m) af[m] = frag_load_nt(a0 + m * 16 * KPAD + k0);
#pragma unroll
    for (int t = 0; t < NT; ++t) bf[t] = frag_load(b0 + t * 16 * KPAD + k0);
#pragma unroll
    for (int m = 0; m < MT; ++m)
#pragma unroll
      for (int t = 0; t < NT; ++t)
        acc[m][t] = __builtin_amdgcn_wmma_f32_16x16x32_f16(false, af[m], false, bf[t],
                                                           (short)0, acc[m][t], false, false);
  }
#pragma unroll
  for (int m = 0; m < MT; ++m) {
#pragma unroll
    for (int t = 0; t < NT; ++t) {
      int col = ct * (NT * 16) + t * 16 + nloc;
      float bv = bias[col];
      float* op = O + ((size_t)((mg * MT + m) * 16 + hi * 8)) * Npad + col;
#pragma unroll
      for (int r = 0; r < 8; ++r) {
        float x = acc[m][t][r] + bv;
        if (act) x = LEAKY(x);
        op[(size_t)r * Npad] = x;
      }
    }
  }
}

// ---------------------------------------------------------------------------
// WMMA GEMM with fused bias+leaky+K-neighbor sum. Each 16-row tile = 2
// points of 8 edges. C-layout: VGPR r = row r+8*hi -> lanes 0-15 hold rows
// 0..7 (first point), lanes 16-31 rows 8..15 (second point). Summing the 8
// accumulator VGPRs in-lane performs sum_k for free.
// ---------------------------------------------------------------------------
template <int MT, int NT, int KPAD>
__global__ void wmma_gemm_ksum(const h16* __restrict__ A, const h16* __restrict__ Bt,
                               const float* __restrict__ bias, float* __restrict__ O,
                               int M, int Npad, int strips) {
  int wave = blockIdx.x * (blockDim.x >> 5) + (threadIdx.x >> 5);
  int mgroups = M / (16 * MT);
  if (wave >= mgroups * strips) return;
  int mg = wave / strips, ct = wave % strips;
  int lane = threadIdx.x & 31;
  int nloc = lane & 15;
  int hi = lane >> 4;
  int kb = hi ? 8 : 0;

  const h16* a0 = A + ((size_t)(mg * MT * 16 + nloc)) * KPAD + kb;
  const h16* b0 = Bt + ((size_t)(ct * (NT * 16) + nloc)) * KPAD + kb;
  v8f acc[MT][NT];
#pragma unroll
  for (int m = 0; m < MT; ++m)
#pragma unroll
    for (int t = 0; t < NT; ++t) acc[m][t] = (v8f){};

  for (int k0 = 0; k0 < KPAD; k0 += 32) {
    v16h af[MT], bf[NT];
#pragma unroll
    for (int m = 0; m < MT; ++m) af[m] = frag_load_nt(a0 + m * 16 * KPAD + k0);
#pragma unroll
    for (int t = 0; t < NT; ++t) bf[t] = frag_load(b0 + t * 16 * KPAD + k0);
#pragma unroll
    for (int m = 0; m < MT; ++m)
#pragma unroll
      for (int t = 0; t < NT; ++t)
        acc[m][t] = __builtin_amdgcn_wmma_f32_16x16x32_f16(false, af[m], false, bf[t],
                                                           (short)0, acc[m][t], false, false);
  }
#pragma unroll
  for (int m = 0; m < MT; ++m) {
    size_t point = (size_t)(mg * MT + m) * 2 + hi;
#pragma unroll
    for (int t = 0; t < NT; ++t) {
      int col = ct * (NT * 16) + t * 16 + nloc;
      float bv = bias[col];
      float s = 0.f;
#pragma unroll
      for (int r = 0; r < 8; ++r) {
        float x = acc[m][t][r] + bv;
        s += LEAKY(x);
      }
      O[point * Npad + col] = s;
    }
  }
}

// ---------------------------------------------------------------------------
// Per-(batch,col) pooling over 2048 points: max/min/sum/mean, leaky applied.
// ---------------------------------------------------------------------------
__global__ void pool_kernel(const float* __restrict__ H, float* __restrict__ pooled,
                            int nper, int ncol) {
  __shared__ float smax[256], smin[256], ssum[256];
  int b = blockIdx.x / ncol;
  int col = blockIdx.x % ncol;
  float mx = -3.4e38f, mn = 3.4e38f, sm = 0.f;
  for (int r = threadIdx.x; r < nper; r += blockDim.x) {
    float v = H[((size_t)b * nper + r) * ncol + col];
    mx = fmaxf(mx, v);
    mn = fminf(mn, v);
    sm += v;
  }
  smax[threadIdx.x] = mx; smin[threadIdx.x] = mn; ssum[threadIdx.x] = sm;
  __syncthreads();
  for (int s = 128; s > 0; s >>= 1) {
    if (threadIdx.x < s) {
      smax[threadIdx.x] = fmaxf(smax[threadIdx.x], smax[threadIdx.x + s]);
      smin[threadIdx.x] = fminf(smin[threadIdx.x], smin[threadIdx.x + s]);
      ssum[threadIdx.x] += ssum[threadIdx.x + s];
    }
    __syncthreads();
  }
  if (threadIdx.x == 0) {
    float* o = pooled + (size_t)b * 4 * ncol;
    float a0 = smax[0], a1 = smin[0], a2 = ssum[0], a3 = ssum[0] / (float)nper;
    o[col] = LEAKY(a0);
    o[ncol + col] = LEAKY(a1);
    o[2 * ncol + col] = LEAKY(a2);
    o[3 * ncol + col] = LEAKY(a3);
  }
}

// ---------------------------------------------------------------------------
// Head: out[b] = (leaky(pooled @ n3_w + n3_b)) @ n4_w + n4_b   (tiny)
// ---------------------------------------------------------------------------
__global__ void head_kernel(const float* __restrict__ pooled, const float* __restrict__ n3w,
                            const float* __restrict__ n3b, const float* __restrict__ n4w,
                            const float* __restrict__ n4b, float* __restrict__ out) {
  __shared__ float t[96];
  int b = blockIdx.x;
  int j = threadIdx.x;
  if (j < 96) {
    float s = n3b[j];
    const float* p = pooled + (size_t)b * 768;
    for (int c = 0; c < 768; ++c) s += p[c] * n3w[(size_t)c * 96 + j];
    t[j] = LEAKY(s);
  }
  __syncthreads();
  if (j == 0) {
    float s = n4b[0];
    for (int q = 0; q < 96; ++q) s += t[q] * n4w[q];
    out[b] = s;
  }
}

// ---------------------------------------------------------------------------
extern "C" void kernel_launch(void* const* d_in, const int* in_sizes, int n_in,
                              void* d_out, int out_size, void* d_ws, size_t ws_size,
                              hipStream_t stream) {
  (void)in_sizes; (void)n_in; (void)out_size; (void)ws_size;
  const int B = 16, NPTS = 2048;
  const long long NP = (long long)B * NPTS;   // 32768 points
  const long long NE = NP * 8;                // 262144 edges

  const float* x = (const float*)d_in[0];
  const float* n3w = (const float*)d_in[21];
  const float* n3b = (const float*)d_in[22];
  const float* n4w = (const float*)d_in[23];
  const float* n4b = (const float*)d_in[24];
  float* out = (float*)d_out;

  char* w = (char*)d_ws;
  size_t off = 0;
  auto alloc = [&](size_t bytes) -> char* {
    char* p = w + off;
    off = (off + bytes + 255) & ~(size_t)255;
    return p;
  };

  int* idxb   = (int*)alloc(NE * 4);
  float* ptA  = (float*)alloc(NP * 192 * 4);
  float* ptB  = (float*)alloc(NP * 192 * 4);
  float* ptC  = (float*)alloc(NP * 192 * 4);
  float* ptD  = (float*)alloc(NP * 192 * 4);
  h16* E      = (h16*)alloc(NE * 384 * 2);     // also reused for concat matrix
  h16* H1     = (h16*)alloc(NE * 256 * 2);     // also reused for head hidden
  float* pooled = (float*)alloc(16 * 768 * 4);

  // padded f16 weights (N-major) + padded f32 biases
  h16* c1w1t = (h16*)alloc(96 * 32 * 2);    float* c1b1 = (float*)alloc(96 * 4);
  h16* c1w2t = (h16*)alloc(192 * 96 * 2);   float* c1b2 = (float*)alloc(192 * 4);
  h16* cw1t[3]; float* cb1[3]; h16* cw2t[3]; float* cb2[3];
  for (int l = 0; l < 3; ++l) {
    cw1t[l] = (h16*)alloc(256 * 384 * 2);   cb1[l] = (float*)alloc(256 * 4);
    cw2t[l] = (h16*)alloc(192 * 256 * 2);   cb2[l] = (float*)alloc(192 * 4);
  }
  h16* n1wt = (h16*)alloc(256 * 800 * 2);   float* n1bp = (float*)alloc(256 * 4);
  h16* n2wt = (h16*)alloc(192 * 256 * 2);   float* n2bp = (float*)alloc(192 * 4);

  h16* concatM = E;                               // 52.4 MB <= 201 MB
  h16* Hn = H1;                                   // 16.8 MB
  float* Hf = (float*)((char*)H1 + (64ll << 20)); // 25.2 MB at +64 MB, fits in 128 MB

  auto prep = [&](const float* W, const float* b, h16* Wt, float* bp,
                  int K, int N, int Kpad, int Npad) {
    size_t tot = (size_t)Npad * Kpad;
    prep_weight<<<(unsigned)((tot + 255) / 256), 256, 0, stream>>>(W, b, Wt, bp, K, N, Kpad, Npad);
  };
  prep((const float*)d_in[1],  (const float*)d_in[2],  c1w1t, c1b1, 16, 96, 32, 96);
  prep((const float*)d_in[3],  (const float*)d_in[4],  c1w2t, c1b2, 96, 192, 96, 192);
  for (int l = 0; l < 3; ++l) {
    prep((const float*)d_in[5 + 4 * l], (const float*)d_in[6 + 4 * l], cw1t[l], cb1[l], 384, 252, 384, 256);
    prep((const float*)d_in[7 + 4 * l], (const float*)d_in[8 + 4 * l], cw2t[l], cb2[l], 252, 192, 256, 192);
  }
  prep((const float*)d_in[17], (const float*)d_in[18], n1wt, n1bp, 776, 252, 800, 256);
  prep((const float*)d_in[19], (const float*)d_in[20], n2wt, n2bp, 252, 192, 256, 192);

  auto blocks_for = [](long long M, int MT, int strips) -> unsigned {
    long long waves = (M / (16 * MT)) * strips;
    return (unsigned)((waves + 7) / 8);  // 8 waves per 256-thread block
  };

  // ---- conv1: F=8 (ldE=32), hidden Npad=96 (MT=2,NT=3, 2 strips) ----------
  knn_kernel<<<dim3(NPTS / 256, B), 256, 0, stream>>>(x, idxb, NPTS, 8);
  build_edge<<<(unsigned)(((size_t)NE * 32 + 255) / 256), 256, 0, stream>>>(
      x, idxb, E, NPTS, 8, 32, NE);
  wmma_gemm_f16out<2, 3, 32><<<blocks_for(NE, 2, 2), 256, 0, stream>>>(
      E, c1w1t, c1b1, H1, (int)NE, 96, 2, 1);
  wmma_gemm_ksum<2, 4, 96><<<blocks_for(NE, 2, 3), 256, 0, stream>>>(
      H1, c1w2t, c1b2, ptA, (int)NE, 192, 3);

  // ---- conv2..4: F=192 (ldE=384), hidden Npad=256 (MT=2,NT=4, 4 strips) ---
  const float* pin[3] = { ptA, ptB, ptC };
  float* pout[3] = { ptB, ptC, ptD };
  for (int l = 0; l < 3; ++l) {
    knn_kernel<<<dim3(NPTS / 256, B), 256, 0, stream>>>(pin[l], idxb, NPTS, 192);
    build_edge<<<(unsigned)(((size_t)NE * 384 + 255) / 256), 256, 0, stream>>>(
        pin[l], idxb, E, NPTS, 192, 384, NE);
    wmma_gemm_f16out<2, 4, 384><<<blocks_for(NE, 2, 4), 256, 0, stream>>>(
        E, cw1t[l], cb1[l], H1, (int)NE, 256, 4, 1);
    wmma_gemm_ksum<2, 4, 256><<<blocks_for(NE, 2, 3), 256, 0, stream>>>(
        H1, cw2t[l], cb2[l], pout[l], (int)NE, 192, 3);
  }

  // ---- head MLP -----------------------------------------------------------
  build_concat<<<(unsigned)(((size_t)NP * 800 + 255) / 256), 256, 0, stream>>>(
      x, ptA, ptB, ptC, ptD, concatM, 800, NP);
  wmma_gemm_f16out<2, 4, 800><<<blocks_for(NP, 2, 4), 256, 0, stream>>>(
      concatM, n1wt, n1bp, Hn, (int)NP, 256, 4, 1);
  wmma_gemm_f32out<2, 4, 256><<<blocks_for(NP, 2, 3), 256, 0, stream>>>(
      Hn, n2wt, n2bp, Hf, (int)NP, 192, 3, 0);

  pool_kernel<<<16 * 192, 256, 0, stream>>>(Hf, pooled, NPTS, 192);
  head_kernel<<<16, 128, 0, stream>>>(pooled, n3w, n3b, n4w, n4b, out);
}
